// Block_1606317768885
// MI455X (gfx1250) — matrix operand order repaired and checked by reading, not compiled, not run on hardware
//
#include <hip/hip_runtime.h>
#include <stdint.h>

#define TDIM 64
#define CDIM 384
#define HDIM 64
#define FDIM 1536

typedef __bf16 v16bf __attribute__((ext_vector_type(16)));
typedef __bf16 v8bf  __attribute__((ext_vector_type(8)));
typedef float  v8f   __attribute__((ext_vector_type(8)));

__device__ __forceinline__ __bf16 f2bf(float f) {
  union { float f; uint32_t u; } c; c.f = f;
  return __builtin_bit_cast(__bf16, (uint16_t)(c.u >> 16));
}

__device__ __forceinline__ v8f zero8() {
  v8f z = {0.f, 0.f, 0.f, 0.f, 0.f, 0.f, 0.f, 0.f};
  return z;
}

__device__ __forceinline__ v8f wmma_bf16(v16bf a, v16bf b, v8f c) {
  // D = A(16x32) * B(32x16) + C, fp32 accumulate
  return __builtin_amdgcn_wmma_f32_16x16x32_bf16(false, a, false, b, (short)0, c, false, false);
}

// A fragment 16x32 from row-major bf16 (ld in elements): one contiguous 16-elem run (k = 16*half + s).
__device__ __forceinline__ v16bf load_a(const __bf16* p, int ld) {
  int lane = threadIdx.x & 31;
  const __bf16* row = p + (lane & 15) * ld + (lane >> 4) * 16;
  v8bf lo = *(const v8bf*)(row);
  v8bf hi = *(const v8bf*)(row + 8);
  return __builtin_shufflevector(lo, hi, 0, 1, 2, 3, 4, 5, 6, 7, 8, 9, 10, 11, 12, 13, 14, 15);
}

// B fragment 32x16 from TRANSPOSED ([n][k]) source: one contiguous 32B load per lane.
__device__ __forceinline__ v16bf load_bT(const __bf16* p, int ld) {
  int lane = threadIdx.x & 31;
  return *(const v16bf*)(p + (lane & 15) * ld + (lane >> 4) * 16);
}

// C fragment (f32 16x16) -> row-major bf16. Elem r: m = r + (lane>>4)*8, n = lane&15.
__device__ __forceinline__ void store_c_bf16(__bf16* p, int ld, v8f c) {
  int lane = threadIdx.x & 31;
  int n = lane & 15, mh = (lane >> 4) * 8;
#pragma unroll
  for (int r = 0; r < 8; ++r) p[(mh + r) * ld + n] = c[r];
}

// C fragment -> TRANSPOSED bf16 dest T[n][m]: contiguous in r -> one 16B store.
__device__ __forceinline__ void store_cT_bf16(__bf16* p, int ld, v8f c) {
  int lane = threadIdx.x & 31;
  int n = lane & 15, mh = (lane >> 4) * 8;
  v8bf o;
#pragma unroll
  for (int r = 0; r < 8; ++r) o[r] = f2bf(c[r]);
  *(v8bf*)(p + n * ld + mh) = o;
}

// ---------- prep: batched transpose fp32 -> bf16 (dst[b][n][k] = src[b][k][n]) ----------
__global__ void transpose_to_bf16(const float* __restrict__ src, __bf16* __restrict__ dst,
                                  int K, int N) {
  size_t base = (size_t)blockIdx.y * K * N;
  int idx = blockIdx.x * 256 + threadIdx.x;
  if (idx >= K * N) return;
  int n = idx / K, k = idx - n * K;
  dst[base + (size_t)n * K + k] = f2bf(src[base + (size_t)k * N + n]);
}

// ---------- main fused transformer block ----------
__global__ __launch_bounds__(256) void transformer_block_kernel(
    const float* __restrict__ X, const float* __restrict__ ln1g, const float* __restrict__ ln1b,
    const __bf16* __restrict__ wQ,  // [h][d][c] (transposed per head)
    const __bf16* __restrict__ wK,  // [h][d][c]
    const __bf16* __restrict__ wV,  // [h][d][c]
    const __bf16* __restrict__ wO,  // [cout][cin]
    const float* __restrict__ bo,
    const float* __restrict__ ln2g, const float* __restrict__ ln2b,
    const __bf16* __restrict__ w1,  // [j(1536)][c(384)]
    const float* __restrict__ b1,
    const __bf16* __restrict__ w2,  // [c(384)][j(1536)]
    const float* __restrict__ b2,
    float* __restrict__ Out) {
  // 160 KB LDS arena, regions phase-unioned.
  __shared__ alignas(64) char smem[163840];
  __bf16* sA   = (__bf16*)smem;            // [0,49152)      xn bf16 (64x384)
  __bf16* sK   = (__bf16*)(smem + 49152);  // [49152,65536)  k for 2 heads (64x128)
  __bf16* sVT  = (__bf16*)(smem + 65536);  // [65536,81920)  vT for 2 heads (128x64)
  __bf16* sScr = (__bf16*)(smem + 81920);  // [81920,98304)  per-wave 16x64 scratch
  float*  sXn  = (float*)smem;             // [0,98304)      x after attention (fp32)
  __bf16* sAO  = (__bf16*)(smem + 98304);  // [98304,147456) attn-out bf16, then xn2
  __bf16* sH   = (__bf16*)(smem + 147456); // [147456,163840) ffn hidden chunk 64x128

  const int wave = threadIdx.x >> 5;
  const int lane = threadIdx.x & 31;
  const int b = blockIdx.x;
  const float* xg = X + (size_t)b * TDIM * CDIM;

  // ---- Phase 1: LN1(x) -> sA (bf16). One wave per 8 rows, shfl reductions. ----
  for (int r = wave * 8; r < wave * 8 + 8; ++r) {
    const float* row = xg + r * CDIM;
    float vals[12], s = 0.f, s2 = 0.f;
#pragma unroll
    for (int j = 0; j < 12; ++j) {
      float v = row[lane + j * 32];
      vals[j] = v; s += v; s2 += v * v;
    }
#pragma unroll
    for (int off = 16; off >= 1; off >>= 1) {
      s += __shfl_xor(s, off, 32);
      s2 += __shfl_xor(s2, off, 32);
    }
    float mu = s * (1.f / CDIM);
    float inv = rsqrtf(s2 * (1.f / CDIM) - mu * mu + 1e-5f);
#pragma unroll
    for (int j = 0; j < 12; ++j) {
      int c = lane + j * 32;
      sA[r * CDIM + c] = f2bf((vals[j] - mu) * inv * ln1g[c] + ln1b[c]);
    }
  }
  __syncthreads();

  // ---- Phase 2: attention, two heads per pass (3 passes). ----
  for (int hp = 0; hp < 3; ++hp) {
    // K & V projections: waves 0-3 -> K (rb=wave), waves 4-7 -> V (rb=wave-4).
    // K outer-loop so the A fragment is loaded once per k-step and reused by 8 tiles.
    {
      int mat = wave >> 2;       // 0 = K, 1 = V
      int rb = wave & 3;         // 16-row band
      const __bf16* Wbase = (mat == 0 ? wK : wV);
      v8f acc8[8];
#pragma unroll
      for (int ct = 0; ct < 8; ++ct) acc8[ct] = zero8();
      for (int k0 = 0; k0 < CDIM; k0 += 32) {
        v16bf a = load_a(sA + rb * 16 * CDIM + k0, CDIM);
#pragma unroll
        for (int ct = 0; ct < 8; ++ct) {
          int h = hp * 2 + (ct >> 2);
          acc8[ct] = wmma_bf16(a,
              load_bT(Wbase + h * (HDIM * CDIM) + (ct & 3) * 16 * CDIM + k0, CDIM), acc8[ct]);
        }
      }
#pragma unroll
      for (int ct = 0; ct < 8; ++ct) {
        if (mat == 0)
          store_c_bf16(sK + rb * 16 * 128 + ct * 16, 128, acc8[ct]);   // k[s][d]
        else
          store_cT_bf16(sVT + ct * 16 * 64 + rb * 16, 64, acc8[ct]);   // vT[d][s]
      }
    }
    __syncthreads();

    // Each wave: (head-in-pair, 16-row band) attention tile.
    {
      int hh = wave >> 2, rb = wave & 3, h = hp * 2 + hh;
      __bf16* scr = sScr + wave * (16 * 64);
      // q tile (pre-scaled by 1/sqrt(64)) -> scratch bf16; A hoisted over 4 d-tiles.
      {
        v8f qacc[4];
#pragma unroll
        for (int dt = 0; dt < 4; ++dt) qacc[dt] = zero8();
        for (int k0 = 0; k0 < CDIM; k0 += 32) {
          v16bf a = load_a(sA + rb * 16 * CDIM + k0, CDIM);
#pragma unroll
          for (int dt = 0; dt < 4; ++dt)
            qacc[dt] = wmma_bf16(a,
                load_bT(wQ + h * (HDIM * CDIM) + dt * 16 * CDIM + k0, CDIM), qacc[dt]);
        }
#pragma unroll
        for (int dt = 0; dt < 4; ++dt) {
#pragma unroll
          for (int r = 0; r < 8; ++r) qacc[dt][r] *= 0.125f;
          store_c_bf16(scr + dt * 16, 64, qacc[dt]);
        }
      }
      v16bf qa0 = load_a(scr, 64);
      v16bf qa1 = load_a(scr + 32, 64);
      v8f sc[4];
#pragma unroll
      for (int st = 0; st < 4; ++st) {
        v8f acc = zero8();
        // scores: B(k=d, n=s) = k[s][d] -> transposed source, contiguous reads
        acc = wmma_bf16(qa0, load_bT(sK + st * 16 * 128 + hh * 64, 128), acc);
        acc = wmma_bf16(qa1, load_bT(sK + st * 16 * 128 + hh * 64 + 32, 128), acc);
        sc[st] = acc;
      }
      // causal mask + softmax; row m lives in one VGPR across a 16-lane half.
      int n = lane & 15, mh = (lane >> 4) * 8;
      float mx[8], sm[8];
#pragma unroll
      for (int r = 0; r < 8; ++r) {
        int t = rb * 16 + mh + r;
        float m_ = -1e30f;
#pragma unroll
        for (int st = 0; st < 4; ++st) {
          int s = st * 16 + n;
          float v = sc[st][r];
          if (s > t) { v = -1e30f; sc[st][r] = v; }
          m_ = fmaxf(m_, v);
        }
        mx[r] = m_;
      }
#pragma unroll
      for (int off = 1; off < 16; off <<= 1)
#pragma unroll
        for (int r = 0; r < 8; ++r) mx[r] = fmaxf(mx[r], __shfl_xor(mx[r], off, 32));
#pragma unroll
      for (int r = 0; r < 8; ++r) {
        float s_ = 0.f;
#pragma unroll
        for (int st = 0; st < 4; ++st) {
          float e = __expf(sc[st][r] - mx[r]);
          sc[st][r] = e; s_ += e;
        }
        sm[r] = s_;
      }
#pragma unroll
      for (int off = 1; off < 16; off <<= 1)
#pragma unroll
        for (int r = 0; r < 8; ++r) sm[r] += __shfl_xor(sm[r], off, 32);
#pragma unroll
      for (int r = 0; r < 8; ++r) {
        float inv = 1.f / sm[r];
#pragma unroll
        for (int st = 0; st < 4; ++st) sc[st][r] *= inv;
      }
#pragma unroll
      for (int st = 0; st < 4; ++st) store_c_bf16(scr + st * 16, 64, sc[st]);
      // out tile = att(16x64) @ v(64x64); B(k=s, n=d) = vT[d][s] -> contiguous reads
      v16bf pa0 = load_a(scr, 64);
      v16bf pa1 = load_a(scr + 32, 64);
#pragma unroll
      for (int dt = 0; dt < 4; ++dt) {
        v8f acc = zero8();
        acc = wmma_bf16(pa0, load_bT(sVT + (hh * 64 + dt * 16) * 64, 64), acc);
        acc = wmma_bf16(pa1, load_bT(sVT + (hh * 64 + dt * 16) * 64 + 32, 64), acc);
        store_c_bf16(sAO + rb * 16 * CDIM + h * 64 + dt * 16, CDIM, acc);
      }
    }
    __syncthreads();
  }

  // ---- Phase 3: x = x + attnOut @ Wo + bo -> sXn. A hoisted over 12 col-tiles. ----
  {
    int rb = wave >> 1;
    int ctbase = (wave & 1) * 12;
    v8f acc12[12];
#pragma unroll
    for (int i = 0; i < 12; ++i) acc12[i] = zero8();
    for (int k0 = 0; k0 < CDIM; k0 += 32) {
      v16bf a = load_a(sAO + rb * 16 * CDIM + k0, CDIM);
#pragma unroll
      for (int i = 0; i < 12; ++i)
        acc12[i] = wmma_bf16(a, load_bT(wO + (ctbase + i) * 16 * CDIM + k0, CDIM), acc12[i]);
    }
    int n = lane & 15, mh = (lane >> 4) * 8;
#pragma unroll
    for (int i = 0; i < 12; ++i) {
      int ct = ctbase + i;
#pragma unroll
      for (int r = 0; r < 8; ++r) {
        int t = rb * 16 + mh + r, c = ct * 16 + n;
        sXn[t * CDIM + c] = acc12[i][r] + xg[t * CDIM + c] + bo[c];
      }
    }
  }
  __syncthreads();

  // ---- Phase 4: LN2(sXn) -> sAO reused as xn2 (bf16). ----
  for (int r = wave * 8; r < wave * 8 + 8; ++r) {
    float vals[12], s = 0.f, s2 = 0.f;
#pragma unroll
    for (int j = 0; j < 12; ++j) {
      float v = sXn[r * CDIM + lane + j * 32];
      vals[j] = v; s += v; s2 += v * v;
    }
#pragma unroll
    for (int off = 16; off >= 1; off >>= 1) {
      s += __shfl_xor(s, off, 32);
      s2 += __shfl_xor(s2, off, 32);
    }
    float mu = s * (1.f / CDIM);
    float inv = rsqrtf(s2 * (1.f / CDIM) - mu * mu + 1e-5f);
#pragma unroll
    for (int j = 0; j < 12; ++j) {
      int c = lane + j * 32;
      sAO[r * CDIM + c] = f2bf((vals[j] - mu) * inv * ln2g[c] + ln2b[c]);
    }
  }
  __syncthreads();

  // ---- Phase 5: FFN in 12 chunks of 128 hidden cols; out accumulators in VGPRs. ----
  {
    int rb = wave >> 1;            // 16-row band owned by this wave (all sub-phases)
    int ct12 = (wave & 1) * 12;    // output col-tile base (24 tiles of 16 over 384)
    int ct4 = (wave & 1) * 4;      // hidden col-tile base (8 tiles of 16 over 128)
    v8f acc[12];
#pragma unroll
    for (int i = 0; i < 12; ++i) acc[i] = zero8();
    for (int c0 = 0; c0 < FDIM; c0 += 128) {
      // h = relu(xn2 @ W1[:,c0:c0+128] + b1) -> sH bf16; A hoisted over 4 tiles.
      {
        v8f ha[4];
#pragma unroll
        for (int j = 0; j < 4; ++j) ha[j] = zero8();
        for (int k0 = 0; k0 < CDIM; k0 += 32) {
          v16bf a = load_a(sAO + rb * 16 * CDIM + k0, CDIM);
#pragma unroll
          for (int j = 0; j < 4; ++j)
            ha[j] = wmma_bf16(a,
                load_bT(w1 + (c0 + (ct4 + j) * 16) * CDIM + k0, CDIM), ha[j]);
        }
        int n = lane & 15, mh = (lane >> 4) * 8;
#pragma unroll
        for (int j = 0; j < 4; ++j) {
          int ct = ct4 + j;
#pragma unroll
          for (int r = 0; r < 8; ++r) {
            float v = ha[j][r] + b1[c0 + ct * 16 + n];
            sH[(rb * 16 + mh + r) * 128 + ct * 16 + n] = f2bf(v > 0.f ? v : 0.f);
          }
        }
      }
      __syncthreads();
      // out += h_chunk @ W2[c0:c0+128,:]; A hoisted over 12 tiles.
      for (int k0 = 0; k0 < 128; k0 += 32) {
        v16bf a = load_a(sH + rb * 16 * 128 + k0, 128);
#pragma unroll
        for (int i = 0; i < 12; ++i)
          acc[i] = wmma_bf16(a,
              load_bT(w2 + (ct12 + i) * 16 * FDIM + c0 + k0, FDIM), acc[i]);
      }
      __syncthreads();
    }

    // ---- Final: out = acc + b2 + residual(sXn) -> global. ----
    float* outg = Out + (size_t)b * TDIM * CDIM;
    int n = lane & 15, mh = (lane >> 4) * 8;
#pragma unroll
    for (int i = 0; i < 12; ++i) {
      int ct = ct12 + i;
#pragma unroll
      for (int r = 0; r < 8; ++r) {
        int t = rb * 16 + mh + r, c = ct * 16 + n;
        outg[t * CDIM + c] = acc[i][r] + sXn[t * CDIM + c] + b2[c];
      }
    }
  }
}

extern "C" void kernel_launch(void* const* d_in, const int* in_sizes, int n_in,
                              void* d_out, int out_size, void* d_ws, size_t ws_size,
                              hipStream_t stream) {
  (void)n_in; (void)ws_size; (void)out_size;
  const float* X    = (const float*)d_in[0];
  const float* ln1g = (const float*)d_in[1];
  const float* ln1b = (const float*)d_in[2];
  const float* Wq   = (const float*)d_in[3];
  const float* Wk   = (const float*)d_in[4];
  const float* Wv   = (const float*)d_in[5];
  const float* Wo   = (const float*)d_in[6];
  const float* bo   = (const float*)d_in[7];
  const float* ln2g = (const float*)d_in[8];
  const float* ln2b = (const float*)d_in[9];
  const float* W1   = (const float*)d_in[10];
  const float* b1   = (const float*)d_in[11];
  const float* W2   = (const float*)d_in[12];
  const float* b2   = (const float*)d_in[13];

  // bf16 transposed weight arena in d_ws (~3.5 MB)
  __bf16* wQ = (__bf16*)d_ws;          // 6*64*384   = 147456
  __bf16* wK = wQ + 147456;
  __bf16* wV = wK + 147456;
  __bf16* wO = wV + 147456;            // 384*384    = 147456
  __bf16* w1 = wO + 147456;            // 1536*384   = 589824
  __bf16* w2 = w1 + 589824;            // 384*1536   = 589824

  transpose_to_bf16<<<dim3((384 * 64 + 255) / 256, 6), 256, 0, stream>>>(Wq, wQ, 384, 64);
  transpose_to_bf16<<<dim3((384 * 64 + 255) / 256, 6), 256, 0, stream>>>(Wk, wK, 384, 64);
  transpose_to_bf16<<<dim3((384 * 64 + 255) / 256, 6), 256, 0, stream>>>(Wv, wV, 384, 64);
  transpose_to_bf16<<<dim3((384 * 384 + 255) / 256, 1), 256, 0, stream>>>(Wo, wO, 384, 384);
  transpose_to_bf16<<<dim3((384 * 1536 + 255) / 256, 1), 256, 0, stream>>>(W1, w1, 384, 1536);
  transpose_to_bf16<<<dim3((1536 * 384 + 255) / 256, 1), 256, 0, stream>>>(W2, w2, 1536, 384);

  int B = in_sizes[0] / (TDIM * CDIM);
  hipLaunchKernelGGL(transformer_block_kernel, dim3(B), dim3(256), 0, stream,
                     X, ln1g, ln1b, wQ, wK, wV, wO, bo, ln2g, ln2b, w1, b1, w2, b2,
                     (float*)d_out);
}
